// CGCNNPyGChargeEarlyImproved_74637941670347
// MI455X (gfx1250) — matrix operand order repaired and checked by reading, not compiled, not run on hardware
//
#include <hip/hip_runtime.h>
#include <hip/hip_bf16.h>

// ---------------------------------------------------------------------------
// CDNA5 (gfx1250) CGCNN forward. bf16 WMMA (V_WMMA_F32_16X16X32_BF16) for all
// GEMMs, f32 accumulate, f32 everywhere else. wave32: one wave owns a 16-row
// tile (4 C-fragments of 16x16, 8 VGPRs each). Full-tile uniform fast paths
// avoid per-element exec-mask churn in the store epilogues.
// ---------------------------------------------------------------------------

typedef __bf16 bf16;
typedef __attribute__((ext_vector_type(16))) __bf16 bf16x16;
typedef __attribute__((ext_vector_type(8)))  __bf16 bf16x8;
typedef __attribute__((ext_vector_type(8)))  float  floatx8;

__device__ __forceinline__ float softplus_f(float v) {
    // log1p(exp(v)), stable
    return (v > 20.0f) ? v : log1pf(__expf(v));
}

// A-fragment (16x32 bf16) from LDS, row-major with given element stride.
// Layout per ISA 7.12.2: lanes 0-15 hold row M=lane, K = k0+{0..7} (VGPR0-3)
// and K = k0+{16..23} (VGPR4-7); lanes 16-31 hold K = k0+{8..15} / {24..31}.
__device__ __forceinline__ bf16x16 loadA(const bf16* base, int stride, int k0, int lane) {
    const bf16* p = base + (lane & 15) * stride + k0 + ((lane >> 4) << 3);
    bf16x8 lo = *(const bf16x8*)(p);
    bf16x8 hi = *(const bf16x8*)(p + 16);
    bf16x16 a;
#pragma unroll
    for (int i = 0; i < 8; ++i) { a[i] = lo[i]; a[i + 8] = hi[i]; }
    return a;
}

// B-fragment (32x16 bf16) from LDS weight [K][64] row-major.
// Lane L holds N = n0+(L&15); K = k0+8*(L>>4)+{0..7} and k0+16+8*(L>>4)+{0..7}.
__device__ __forceinline__ bf16x16 loadB(const bf16* w, int k0, int n0, int lane) {
    const bf16* p = w + (size_t)(k0 + ((lane >> 4) << 3)) * 64 + n0 + (lane & 15);
    bf16x16 b;
#pragma unroll
    for (int i = 0; i < 8; ++i) { b[i] = p[i * 64]; b[i + 8] = p[(16 + i) * 64]; }
    return b;
}

#define WMMA_BF16(a, b, c) \
    __builtin_amdgcn_wmma_f32_16x16x32_bf16(false, (a), false, (b), (short)0, (c), false, false)

// ---------------------------------------------------------------------------
// Weight f32 -> bf16 with zero row padding (all matrices have 64 cols).
// ---------------------------------------------------------------------------
__global__ void cvt_pad_kernel(const float* __restrict__ src, bf16* __restrict__ dst,
                               int rows, int padRows) {
    int i = blockIdx.x * 256 + threadIdx.x;
    int total = padRows * 64;
    if (i < total) {
        int r = i >> 6;
        dst[i] = (r < rows) ? (bf16)src[i] : (bf16)0.0f;
    }
}

// ---------------------------------------------------------------------------
// Node embedding: [x | charge_embed(batch)] [N,108->128] @ atom_w -> xbuf[N,64]
// ---------------------------------------------------------------------------
__global__ __launch_bounds__(128)
void node_embed_kernel(const float* __restrict__ x_in, const float* __restrict__ charge,
                       const float* __restrict__ cw, const float* __restrict__ cb,
                       const int* __restrict__ batch, const bf16* __restrict__ gW,
                       const float* __restrict__ gb, float* __restrict__ xbuf, int nN) {
    __shared__ bf16 sW[128 * 64];
    __shared__ float sB[64], sCW[16], sCB[16];
    __shared__ bf16 sA[4][16 * 128];
    const int tid = threadIdx.x, lane = tid & 31, wave = tid >> 5;
    {
        const uint4* s = (const uint4*)gW; uint4* d = (uint4*)sW;
        for (int i = tid; i < 1024; i += 128) d[i] = s[i];
        if (tid < 64) sB[tid] = gb[tid];
        if (tid < 16) { sCW[tid] = cw[tid]; sCB[tid] = cb[tid]; }
    }
    __syncthreads();
    bf16* A = sA[wave];
    const int ntiles = (nN + 15) >> 4;
    for (int t = blockIdx.x * 4 + wave; t < ntiles; t += gridDim.x * 4) {
        const int n0row = t << 4;
        const bool fullTile = (n0row + 16) <= nN;   // wave-uniform
        for (int r = 0; r < 16; ++r) {
            int nid = n0row + r;
            bool ok = nid < nN;                     // uniform per r
            bf16* row = A + r * 128;
            for (int j = lane; j < 128; j += 32) {
                float v = 0.0f;
                if (ok) {
                    if (j < 92) v = x_in[(size_t)nid * 92 + j];
                    else if (j < 108) {
                        int g = batch[nid];
                        v = charge[g] * sCW[j - 92] + sCB[j - 92];
                    }
                }
                row[j] = (bf16)v;
            }
        }
#pragma unroll
        for (int n = 0; n < 4; ++n) {
            floatx8 acc = {0, 0, 0, 0, 0, 0, 0, 0};
#pragma unroll
            for (int kk = 0; kk < 4; ++kk) {
                bf16x16 a = loadA(A, 128, kk * 32, lane);
                bf16x16 b = loadB(sW, kk * 32, n * 16, lane);
                acc = WMMA_BF16(a, b, acc);
            }
            const int cc = n * 16 + (lane & 15);
            const int mb = (lane >> 4) << 3;
            const float bias = sB[cc];
            float* p = xbuf + (size_t)(n0row + mb) * 64 + cc;
            if (fullTile) {
#pragma unroll
                for (int v2 = 0; v2 < 8; ++v2) p[(size_t)v2 * 64] = acc[v2] + bias;
            } else {
#pragma unroll
                for (int v2 = 0; v2 < 8; ++v2)
                    if (n0row + mb + v2 < nN) p[(size_t)v2 * 64] = acc[v2] + bias;
            }
        }
    }
}

// ---------------------------------------------------------------------------
// Edge embedding: edge_attr [E,41->64] @ bond_w -> ebuf[E,64]
// ---------------------------------------------------------------------------
__global__ __launch_bounds__(128)
void edge_embed_kernel(const float* __restrict__ ea, const bf16* __restrict__ gW,
                       const float* __restrict__ gb, float* __restrict__ e_out, int nE) {
    __shared__ bf16 sW[64 * 64];
    __shared__ float sB[64];
    __shared__ bf16 sA[4][16 * 64];
    const int tid = threadIdx.x, lane = tid & 31, wave = tid >> 5;
    {
        const uint4* s = (const uint4*)gW; uint4* d = (uint4*)sW;
        for (int i = tid; i < 512; i += 128) d[i] = s[i];
        if (tid < 64) sB[tid] = gb[tid];
    }
    __syncthreads();
    bf16* A = sA[wave];
    const int ntiles = (nE + 15) >> 4;
    for (int t = blockIdx.x * 4 + wave; t < ntiles; t += gridDim.x * 4) {
        const int e0 = t << 4;
        const bool fullTile = (e0 + 16) <= nE;
        for (int r = 0; r < 16; ++r) {
            int eid = e0 + r;
            bf16* row = A + r * 64;
            for (int j = lane; j < 64; j += 32) {
                float v = 0.0f;
                if (eid < nE && j < 41) v = ea[(size_t)eid * 41 + j];
                row[j] = (bf16)v;
            }
        }
#pragma unroll
        for (int n = 0; n < 4; ++n) {
            floatx8 acc = {0, 0, 0, 0, 0, 0, 0, 0};
#pragma unroll
            for (int kk = 0; kk < 2; ++kk) {
                bf16x16 a = loadA(A, 64, kk * 32, lane);
                bf16x16 b = loadB(sW, kk * 32, n * 16, lane);
                acc = WMMA_BF16(a, b, acc);
            }
            const int cc = n * 16 + (lane & 15);
            const int mb = (lane >> 4) << 3;
            const float bias = sB[cc];
            float* p = e_out + (size_t)(e0 + mb) * 64 + cc;
            if (fullTile) {
#pragma unroll
                for (int v2 = 0; v2 < 8; ++v2) p[(size_t)v2 * 64] = acc[v2] + bias;
            } else {
#pragma unroll
                for (int v2 = 0; v2 < 8; ++v2)
                    if (e0 + mb + v2 < nE) p[(size_t)v2 * 64] = acc[v2] + bias;
            }
        }
    }
}

// ---------------------------------------------------------------------------
// Fused conv layer per 16-edge tile:
//   gather xr,xc,e -> GEMM1(K=192)+softplus -> GEMM2(K=64) -> e (in place)
//   -> GEMM3([xr|e],K=128)+softplus -> GEMM4(K=64) -> atomic scatter into xn
// e in-place is safe: each edge row is read and written only by its own tile.
// ---------------------------------------------------------------------------
__global__ __launch_bounds__(128)
void edge_layer_kernel(const float* __restrict__ x,
                       float* e_buf, /* read+write in place */
                       float* __restrict__ xn,
                       const int* __restrict__ erow, const int* __restrict__ ecol,
                       const bf16* __restrict__ gW1, const bf16* __restrict__ gW2,
                       const bf16* __restrict__ gW3, const bf16* __restrict__ gW4,
                       const float* __restrict__ geb1, const float* __restrict__ geb2,
                       const float* __restrict__ gnb1, const float* __restrict__ gnb2,
                       int nE) {
    __shared__ bf16 sW1[192 * 64];
    __shared__ bf16 sW2[64 * 64];
    __shared__ bf16 sW3[128 * 64];
    __shared__ bf16 sW4[64 * 64];
    __shared__ float sB1[64], sB2[64], sB3[64], sB4[64];
    __shared__ bf16 sEI[4][16 * 192];   // [xr | xc->e_new | e_in] rows, stride 192
    __shared__ bf16 sT[4][16 * 64];     // intermediate softplus tile

    const int tid = threadIdx.x, lane = tid & 31, wave = tid >> 5;
    {
        const uint4* s1 = (const uint4*)gW1; uint4* d1 = (uint4*)sW1;
        for (int i = tid; i < 1536; i += 128) d1[i] = s1[i];
        const uint4* s2 = (const uint4*)gW2; uint4* d2 = (uint4*)sW2;
        for (int i = tid; i < 512; i += 128) d2[i] = s2[i];
        const uint4* s3 = (const uint4*)gW3; uint4* d3 = (uint4*)sW3;
        for (int i = tid; i < 1024; i += 128) d3[i] = s3[i];
        const uint4* s4 = (const uint4*)gW4; uint4* d4 = (uint4*)sW4;
        for (int i = tid; i < 512; i += 128) d4[i] = s4[i];
        if (tid < 64) { sB1[tid] = geb1[tid]; sB2[tid] = geb2[tid];
                        sB3[tid] = gnb1[tid]; sB4[tid] = gnb2[tid]; }
    }
    __syncthreads();

    bf16* EI = sEI[wave];
    bf16* T  = sT[wave];
    const int ntiles = (nE + 15) >> 4;

    for (int t = blockIdx.x * 4 + wave; t < ntiles; t += gridDim.x * 4) {
        const int e0 = t << 4;
        const bool fullTile = (e0 + 16) <= nE;   // wave-uniform
        // --- edge indices (lanes 0..15 hold them; broadcast via shuffle) ---
        int rv = 0, cv = 0;
        if (lane < 16) {
            int eid = e0 + lane;
            if (eid < nE) { rv = erow[eid]; cv = ecol[eid]; }
        }
        // --- gather xr, xc, e into EI (bf16) ---
        for (int r = 0; r < 16; ++r) {
            int eid = e0 + r;
            int ri = __shfl(rv, r);
            int ci = __shfl(cv, r);
            int c = lane * 2;
            float2 vx = {0.f, 0.f}, vc = {0.f, 0.f}, ve = {0.f, 0.f};
            if (eid < nE) {   // uniform per r
                vx = *(const float2*)(x + (size_t)ri * 64 + c);
                vc = *(const float2*)(x + (size_t)ci * 64 + c);
                ve = *(const float2*)(e_buf + (size_t)eid * 64 + c);
            }
            bf16* row = EI + r * 192;
            row[c] = (bf16)vx.x;       row[c + 1] = (bf16)vx.y;
            row[64 + c] = (bf16)vc.x;  row[64 + c + 1] = (bf16)vc.y;
            row[128 + c] = (bf16)ve.x; row[128 + c + 1] = (bf16)ve.y;
        }
        // --- GEMM1: [16,192] @ ew1 -> softplus -> T ---
#pragma unroll
        for (int n = 0; n < 4; ++n) {
            floatx8 acc = {0, 0, 0, 0, 0, 0, 0, 0};
#pragma unroll
            for (int kk = 0; kk < 6; ++kk) {
                bf16x16 a = loadA(EI, 192, kk * 32, lane);
                bf16x16 b = loadB(sW1, kk * 32, n * 16, lane);
                acc = WMMA_BF16(a, b, acc);
            }
            const int cc = n * 16 + (lane & 15);
            const int mb = (lane >> 4) << 3;
            const float bias = sB1[cc];
#pragma unroll
            for (int v2 = 0; v2 < 8; ++v2)
                T[(mb + v2) * 64 + cc] = (bf16)softplus_f(acc[v2] + bias);
        }
        // --- GEMM2: T @ ew2 -> e_new; write global e and EI cols 64..127 ---
#pragma unroll
        for (int n = 0; n < 4; ++n) {
            floatx8 acc = {0, 0, 0, 0, 0, 0, 0, 0};
#pragma unroll
            for (int kk = 0; kk < 2; ++kk) {
                bf16x16 a = loadA(T, 64, kk * 32, lane);
                bf16x16 b = loadB(sW2, kk * 32, n * 16, lane);
                acc = WMMA_BF16(a, b, acc);
            }
            const int cc = n * 16 + (lane & 15);
            const int mb = (lane >> 4) << 3;
            const float bias = sB2[cc];
            float* p = e_buf + (size_t)(e0 + mb) * 64 + cc;
            if (fullTile) {
#pragma unroll
                for (int v2 = 0; v2 < 8; ++v2) {
                    float val = acc[v2] + bias;
                    EI[(mb + v2) * 192 + 64 + cc] = (bf16)val;   // A3 = [xr | e_new]
                    p[(size_t)v2 * 64] = val;
                }
            } else {
#pragma unroll
                for (int v2 = 0; v2 < 8; ++v2) {
                    float val = acc[v2] + bias;
                    EI[(mb + v2) * 192 + 64 + cc] = (bf16)val;
                    if (e0 + mb + v2 < nE) p[(size_t)v2 * 64] = val;
                }
            }
        }
        // --- GEMM3: [xr | e_new] [16,128] @ nw1 -> softplus -> T ---
#pragma unroll
        for (int n = 0; n < 4; ++n) {
            floatx8 acc = {0, 0, 0, 0, 0, 0, 0, 0};
#pragma unroll
            for (int kk = 0; kk < 4; ++kk) {
                bf16x16 a = loadA(EI, 192, kk * 32, lane);
                bf16x16 b = loadB(sW3, kk * 32, n * 16, lane);
                acc = WMMA_BF16(a, b, acc);
            }
            const int cc = n * 16 + (lane & 15);
            const int mb = (lane >> 4) << 3;
            const float bias = sB3[cc];
#pragma unroll
            for (int v2 = 0; v2 < 8; ++v2)
                T[(mb + v2) * 64 + cc] = (bf16)softplus_f(acc[v2] + bias);
        }
        // --- GEMM4: T @ nw2 -> m; atomic scatter-add into xn[col] ---
#pragma unroll
        for (int n = 0; n < 4; ++n) {
            floatx8 acc = {0, 0, 0, 0, 0, 0, 0, 0};
#pragma unroll
            for (int kk = 0; kk < 2; ++kk) {
                bf16x16 a = loadA(T, 64, kk * 32, lane);
                bf16x16 b = loadB(sW4, kk * 32, n * 16, lane);
                acc = WMMA_BF16(a, b, acc);
            }
            const int cc = n * 16 + (lane & 15);
            const int mb = (lane >> 4) << 3;
            const float bias = sB4[cc];
            if (fullTile) {
#pragma unroll
                for (int v2 = 0; v2 < 8; ++v2) {
                    int ci = __shfl(cv, mb + v2);
                    atomicAdd(xn + (size_t)ci * 64 + cc, acc[v2] + bias);
                }
            } else {
#pragma unroll
                for (int v2 = 0; v2 < 8; ++v2) {
                    int m = mb + v2;
                    int ci = __shfl(cv, m);
                    if (e0 + m < nE)
                        atomicAdd(xn + (size_t)ci * 64 + cc, acc[v2] + bias);
                }
            }
        }
    }
}

// ---------------------------------------------------------------------------
// BatchNorm statistics: stats[0..63]=sum, stats[64..127]=sumsq over N rows
// ---------------------------------------------------------------------------
__global__ __launch_bounds__(256)
void bn_stats_kernel(const float* __restrict__ xn, float* __restrict__ stats, int nN) {
    const int c = threadIdx.x & 63;
    const int rl = threadIdx.x >> 6;
    float s = 0.f, s2 = 0.f;
    for (int r = blockIdx.x * 4 + rl; r < nN; r += gridDim.x * 4) {
        float v = xn[(size_t)r * 64 + c];
        s += v; s2 += v * v;
    }
    __shared__ float red[256], red2[256];
    red[threadIdx.x] = s; red2[threadIdx.x] = s2;
    __syncthreads();
    if (rl == 0) {
        s  = red[c]  + red[64 + c]  + red[128 + c]  + red[192 + c];
        s2 = red2[c] + red2[64 + c] + red2[128 + c] + red2[192 + c];
        atomicAdd(&stats[c], s);
        atomicAdd(&stats[64 + c], s2);
    }
}

__global__ __launch_bounds__(256)
void bn_apply_kernel(const float* __restrict__ xn, const float* __restrict__ stats,
                     const float* __restrict__ g, const float* __restrict__ b,
                     float* __restrict__ x, int nN) {
    size_t idx = (size_t)blockIdx.x * 256 + threadIdx.x;
    size_t total = (size_t)nN * 64;
    if (idx >= total) return;
    int c = (int)(idx & 63);
    float invN = 1.0f / (float)nN;
    float mean = stats[c] * invN;
    float var = stats[64 + c] * invN - mean * mean;
    float rstd = rsqrtf(var + 1e-5f);
    float v = (xn[idx] - mean) * rstd * g[c] + b[c];
    x[idx] = softplus_f(v) + x[idx];
}

// ---------------------------------------------------------------------------
// Global mean pool (sum + count via atomics)
// ---------------------------------------------------------------------------
__global__ __launch_bounds__(256)
void pool_kernel(const float* __restrict__ x, const int* __restrict__ batch,
                 float* __restrict__ gsum, float* __restrict__ gcnt, int nN) {
    size_t idx = (size_t)blockIdx.x * 256 + threadIdx.x;
    size_t total = (size_t)nN * 64;
    if (idx >= total) return;
    int n = (int)(idx >> 6);
    int c = (int)(idx & 63);
    int g = batch[n];
    atomicAdd(&gsum[(size_t)g * 64 + c], x[idx]);
    if (c == 0) atomicAdd(&gcnt[g], 1.0f);
}

// ---------------------------------------------------------------------------
// Head MLP: g[G,64] -> 128 -> 128 -> 1  (single block, f32)
// ---------------------------------------------------------------------------
__global__ __launch_bounds__(128)
void head_kernel(const float* __restrict__ gsum, const float* __restrict__ gcnt,
                 const float* __restrict__ pw1, const float* __restrict__ pb1,
                 const float* __restrict__ pw2, const float* __restrict__ pb2,
                 const float* __restrict__ pw3, const float* __restrict__ pb3,
                 float* __restrict__ out, int nG) {
    __shared__ float h1[64 * 128];
    __shared__ float red[128];
    const int tid = threadIdx.x;
    for (int i = tid; i < nG * 128; i += 128) {
        int row = i >> 7, col = i & 127;
        float cnt = fmaxf(gcnt[row], 1.0f);
        float acc = pb1[col];
        for (int kk = 0; kk < 64; ++kk)
            acc += (gsum[(size_t)row * 64 + kk] / cnt) * pw1[kk * 128 + col];
        h1[i] = softplus_f(acc);
    }
    __syncthreads();
    const int row = tid >> 1, half = tid & 1;
    float part = 0.0f;
    if (row < nG) {
        for (int j = half * 64; j < half * 64 + 64; ++j) {
            float acc = pb2[j];
            for (int kk = 0; kk < 128; ++kk)
                acc += h1[row * 128 + kk] * pw2[kk * 128 + j];
            part += softplus_f(acc) * pw3[j];
        }
    }
    red[tid] = part;
    __syncthreads();
    if (half == 0 && row < nG) out[row] = red[tid] + red[tid + 1] + pb3[0];
}

// ---------------------------------------------------------------------------
// Host orchestration
// ---------------------------------------------------------------------------
extern "C" void kernel_launch(void* const* d_in, const int* in_sizes, int n_in,
                              void* d_out, int out_size, void* d_ws, size_t ws_size,
                              hipStream_t stream) {
    (void)n_in; (void)out_size; (void)ws_size;
    const int N = in_sizes[0] / 92;
    const int E = in_sizes[1] / 41;
    const int G = in_sizes[2];

    // --- input pointers (insertion-order pytree flatten of setup_inputs) ---
    const float* x_in = (const float*)d_in[0];
    const float* ea   = (const float*)d_in[1];
    const float* chg  = (const float*)d_in[2];
    int k = 3;
    const float* charge_w = (const float*)d_in[k++];
    const float* charge_b = (const float*)d_in[k++];
    const float* atom_w   = (const float*)d_in[k++];
    const float* atom_b   = (const float*)d_in[k++];
    const float* bond_w   = (const float*)d_in[k++];
    const float* bond_b   = (const float*)d_in[k++];
    const float *ew1[3], *eb1[3], *ew2[3], *eb2[3], *nw1[3], *nb1[3], *nw2[3], *nb2[3], *bng[3], *bnb[3];
    for (int l = 0; l < 3; ++l) {
        ew1[l] = (const float*)d_in[k++]; eb1[l] = (const float*)d_in[k++];
        ew2[l] = (const float*)d_in[k++]; eb2[l] = (const float*)d_in[k++];
        nw1[l] = (const float*)d_in[k++]; nb1[l] = (const float*)d_in[k++];
        nw2[l] = (const float*)d_in[k++]; nb2[l] = (const float*)d_in[k++];
        bng[l] = (const float*)d_in[k++]; bnb[l] = (const float*)d_in[k++];
    }
    const float* pw1 = (const float*)d_in[k++];
    const float* pb1 = (const float*)d_in[k++];
    const float* pw2 = (const float*)d_in[k++];
    const float* pb2 = (const float*)d_in[k++];
    const float* pw3 = (const float*)d_in[k++];
    const float* pb3 = (const float*)d_in[k++];
    const int* edge_index = (const int*)d_in[k++];
    const int* batch      = (const int*)d_in[k++];
    const int* erow = edge_index;
    const int* ecol = edge_index + E;

    // --- workspace carve-up ---
    size_t off = 0;
    char* wsp = (char*)d_ws;
    auto take = [&](size_t bytes) -> void* {
        void* p = wsp + off;
        off += (bytes + 255) & ~(size_t)255;
        return p;
    };
    bf16* AW = (bf16*)take((size_t)128 * 64 * 2);
    bf16* BW = (bf16*)take((size_t)64 * 64 * 2);
    bf16 *W1[3], *W2[3], *W3[3], *W4[3];
    for (int l = 0; l < 3; ++l) {
        W1[l] = (bf16*)take((size_t)192 * 64 * 2);
        W2[l] = (bf16*)take((size_t)64 * 64 * 2);
        W3[l] = (bf16*)take((size_t)128 * 64 * 2);
        W4[l] = (bf16*)take((size_t)64 * 64 * 2);
    }
    float* xbuf  = (float*)take((size_t)N * 64 * 4);
    float* ebuf  = (float*)take((size_t)E * 64 * 4);
    float* xnbuf = (float*)take((size_t)N * 64 * 4);
    float* stats = (float*)take(128 * 4);
    float* gsum  = (float*)take((size_t)G * 64 * 4);
    float* gcnt  = (float*)take((size_t)G * 4);

    // --- weight conversion to bf16 (zero-padded rows) ---
    auto cvt = [&](const float* src, bf16* dst, int rows, int padRows) {
        int total = padRows * 64;
        cvt_pad_kernel<<<(total + 255) / 256, 256, 0, stream>>>(src, dst, rows, padRows);
    };
    cvt(atom_w, AW, 108, 128);
    cvt(bond_w, BW, 41, 64);
    for (int l = 0; l < 3; ++l) {
        cvt(ew1[l], W1[l], 192, 192);
        cvt(ew2[l], W2[l], 64, 64);
        cvt(nw1[l], W3[l], 128, 128);
        cvt(nw2[l], W4[l], 64, 64);
    }

    // --- embeddings ---
    const int ntN = (N + 15) / 16;
    const int ntE = (E + 15) / 16;
    node_embed_kernel<<<(ntN + 3) / 4, 128, 0, stream>>>(
        x_in, chg, charge_w, charge_b, batch, AW, atom_b, xbuf, N);
    edge_embed_kernel<<<(ntE + 3) / 4, 128, 0, stream>>>(ea, BW, bond_b, ebuf, E);

    // --- conv layers ---
    for (int l = 0; l < 3; ++l) {
        hipMemsetAsync(xnbuf, 0, (size_t)N * 64 * 4, stream);
        hipMemsetAsync(stats, 0, 128 * 4, stream);
        edge_layer_kernel<<<(ntE + 3) / 4, 128, 0, stream>>>(
            xbuf, ebuf, xnbuf, erow, ecol,
            W1[l], W2[l], W3[l], W4[l],
            eb1[l], eb2[l], nb1[l], nb2[l], E);
        bn_stats_kernel<<<256, 256, 0, stream>>>(xnbuf, stats, N);
        bn_apply_kernel<<<(int)(((size_t)N * 64 + 255) / 256), 256, 0, stream>>>(
            xnbuf, stats, bng[l], bnb[l], xbuf, N);
    }

    // --- pool + head ---
    hipMemsetAsync(gsum, 0, (size_t)G * 64 * 4, stream);
    hipMemsetAsync(gcnt, 0, (size_t)G * 4, stream);
    pool_kernel<<<(int)(((size_t)N * 64 + 255) / 256), 256, 0, stream>>>(
        xbuf, batch, gsum, gcnt, N);
    head_kernel<<<1, 128, 0, stream>>>(gsum, gcnt, pw1, pb1, pw2, pb2, pw3, pb3,
                                       (float*)d_out, G);
}
